// PluckerEncoder_88811333747374
// MI455X (gfx1250) — matrix operand order repaired and checked by reading, not compiled, not run on hardware
//
#include <hip/hip_runtime.h>
#include <stdint.h>

// ---------------- problem constants (from reference setup_inputs) ----------
#define B_BATCH 8
#define L_SEQ   4096
#define D_MODEL 1024
#define D_RED   32
#define PAIRS   496                      // D_RED*(D_RED-1)/2
#define P_PAD   512                      // K padded to WMMA multiple
#define M_TOT   (B_BATCH * L_SEQ)        // 32768 rows

// ---------------- CDNA5 WMMA types ----------------------------------------
typedef __attribute__((ext_vector_type(16))) __bf16 v16bf;
typedef __attribute__((ext_vector_type(8)))  float  v8f;

union Frag {
  v16bf v;
  uint4 q[2];
  unsigned short s[16];
};

__device__ __forceinline__ v8f wmma_bf16(v16bf a, v16bf b, v8f c) {
  // v_wmma_f32_16x16x32_bf16  D = A(16x32) * B(32x16) + C
  return __builtin_amdgcn_wmma_f32_16x16x32_bf16(
      /*neg_a=*/false, a, /*neg_b=*/false, b,
      /*c_mod=*/(short)0, c, /*reuse_a=*/false, /*reuse_b=*/false);
}

__device__ __forceinline__ unsigned short f2bf(float f) {
  unsigned int u = __float_as_uint(f);
  u += 0x7FFFu + ((u >> 16) & 1u);   // round-to-nearest-even
  return (unsigned short)(u >> 16);
}
__device__ __forceinline__ float bf2f(unsigned short s) {
  return __uint_as_float(((unsigned int)s) << 16);
}

// ---------------- Tensor Data Mover (TDM) ----------------------------------
#if defined(__has_builtin)
#if __has_builtin(__builtin_amdgcn_tensor_load_to_lds) && \
    __has_builtin(__builtin_amdgcn_s_wait_tensorcnt)
#define HAVE_TDM 1
#endif
#endif
#ifndef HAVE_TDM
#define HAVE_TDM 0
#endif

#if HAVE_TDM
typedef unsigned int u32x4 __attribute__((ext_vector_type(4)));
typedef int          i32x4 __attribute__((ext_vector_type(4)));
typedef int          i32x8 __attribute__((ext_vector_type(8)));

__device__ __forceinline__ unsigned int lds_offset(const void* p) {
  // LDS aperture: flat address low 32 bits == LDS byte offset
  return (unsigned int)(uintptr_t)p;
}

// 2D tile load, D# packed per CDNA5 ISA.
// dsz_code: 0=1B,1=2B,2=4B,3=8B. Dims/strides in element units.
__device__ __forceinline__ void tdm_load_2d(unsigned int lds_off, const void* gptr,
                                            unsigned int dsz_code,
                                            unsigned int tile_d0, unsigned int tile_d1,
                                            unsigned int tensor_d0, unsigned int tensor_d1,
                                            unsigned long long stride0) {
  unsigned long long ga = (unsigned long long)(uintptr_t)gptr;
  u32x4 g0;
  g0[0] = 1u;                                                  // count=1, user D#
  g0[1] = lds_off;                                             // lds_addr [63:32]
  g0[2] = (unsigned int)(ga & 0xFFFFFFFFu);                    // global_addr lo
  g0[3] = (unsigned int)((ga >> 32) & 0x01FFFFFFu) | (2u << 30); // addr hi | type=2
  i32x8 g1;
  g1[0] = (int)((dsz_code & 3u) << 16);                        // data_size [17:16]
  g1[1] = (int)((tensor_d0 & 0xFFFFu) << 16);                  // tensor_dim0 lo16 @48
  g1[2] = (int)(((tensor_d0 >> 16) & 0xFFFFu) |
                ((tensor_d1 & 0xFFFFu) << 16));                // dim0 hi | dim1 lo
  g1[3] = (int)(((tensor_d1 >> 16) & 0xFFFFu) |
                ((tile_d0 & 0xFFFFu) << 16));                  // dim1 hi | tile_dim0
  g1[4] = (int)(tile_d1 & 0xFFFFu);                            // tile_dim1 (tile_dim2=0)
  g1[5] = (int)(stride0 & 0xFFFFFFFFull);                      // dim0_stride lo32
  g1[6] = (int)((stride0 >> 32) & 0xFFFFull);                  // dim0_stride hi16
  g1[7] = 0;
  i32x4 g2 = {0, 0, 0, 0};
  i32x4 g3 = {0, 0, 0, 0};
#if __clang_major__ >= 23
  i32x8 g4 = {0, 0, 0, 0, 0, 0, 0, 0};
  __builtin_amdgcn_tensor_load_to_lds(g0, g1, g2, g3, g4, 0);
#else
  __builtin_amdgcn_tensor_load_to_lds(g0, g1, g2, g3, 0);
#endif
}
#endif // HAVE_TDM

// ---------------------------------------------------------------------------
// Kernel 0: weight prep.
//   wredT[r][k]  = bf16(W_red_w[k][r])                      (32 x 1024)
//   wpT[n][p]    = bf16(W_plu_w[p][n]), p>=496 -> 0         (1024 x 512)
// Transposed so WMMA B-fragments read 16 contiguous K values per lane.
// ---------------------------------------------------------------------------
__global__ __launch_bounds__(256) void k_prep(const float* __restrict__ wred,
                                              const float* __restrict__ wplu,
                                              unsigned short* __restrict__ wredT,
                                              unsigned short* __restrict__ wpT) {
  int idx = blockIdx.x * 256 + threadIdx.x;
  if (idx < D_RED * D_MODEL) {
    int r = idx / D_MODEL, k = idx % D_MODEL;
    wredT[idx] = f2bf(wred[(size_t)k * D_RED + r]);
  }
  if (idx < D_MODEL * P_PAD) {
    int n = idx / P_PAD, p = idx % P_PAD;
    wpT[idx] = (p < PAIRS) ? f2bf(wplu[(size_t)p * D_MODEL + n]) : (unsigned short)0;
  }
}

// ---------------------------------------------------------------------------
// Kernel 1: z = h @ W_red + b   (M=32768, N=32, K=1024), bf16 WMMA, f32 out.
// Block = 256 threads (8 waves), 128-row M tile (16 rows/wave).
// h tile (128x32 f32) staged to LDS per K-step via TDM.
// ---------------------------------------------------------------------------
__global__ __launch_bounds__(256) void k_gemm1(const float* __restrict__ h,
                                               const unsigned short* __restrict__ wredT,
                                               const float* __restrict__ bred,
                                               float* __restrict__ z) {
  __shared__ float tile[128 * 32];          // 16 KB
  const int tid  = threadIdx.x;
  const int wave = tid >> 5, lane = tid & 31;
  const int half = lane >> 4, nl = lane & 15;
  const int m0   = blockIdx.x * 128;

  v8f c0 = {}, c1 = {};
  for (int k0 = 0; k0 < D_MODEL; k0 += 32) {
    __syncthreads();                        // tile consumers of prev step done
#if HAVE_TDM
    if (wave == 0) {
      tdm_load_2d(lds_offset(tile), h + (size_t)m0 * D_MODEL + k0,
                  /*4B*/2, /*tile*/32, 128, /*tensor*/D_MODEL, M_TOT,
                  /*stride0*/(unsigned long long)D_MODEL);
      __builtin_amdgcn_s_wait_tensorcnt(0);
    }
#else
    for (int t = tid; t < 128 * 8; t += 256) {
      int r = t >> 3, cq = t & 7;
      ((float4*)tile)[t] = *(const float4*)&h[(size_t)(m0 + r) * D_MODEL + k0 + cq * 4];
    }
#endif
    __syncthreads();

    // A fragment (16x32 bf16): lanes 0-15 -> K 0..15; 16-31 -> K 16..31
    Frag a;
    const float* row = &tile[(wave * 16 + nl) * 32];
#pragma unroll
    for (int g = 0; g < 2; ++g) {
      const float* src = row + half * 16 + g * 8;
      float4 f0 = *(const float4*)src;
      float4 f1 = *(const float4*)(src + 4);
      a.s[g * 8 + 0] = f2bf(f0.x); a.s[g * 8 + 1] = f2bf(f0.y);
      a.s[g * 8 + 2] = f2bf(f0.z); a.s[g * 8 + 3] = f2bf(f0.w);
      a.s[g * 8 + 4] = f2bf(f1.x); a.s[g * 8 + 5] = f2bf(f1.y);
      a.s[g * 8 + 6] = f2bf(f1.z); a.s[g * 8 + 7] = f2bf(f1.w);
    }
    // B fragments (32x16): lane = column, lanes 0-15 K 0-15, 16-31 K 16-31.
    // wredT is [n][k] so the 16 K values are contiguous.
    Frag b0, b1;
    const unsigned short* w0 = &wredT[(size_t)nl * D_MODEL + k0 + half * 16];
    b0.q[0] = *(const uint4*)w0;       b0.q[1] = *(const uint4*)(w0 + 8);
    const unsigned short* w1 = &wredT[(size_t)(16 + nl) * D_MODEL + k0 + half * 16];
    b1.q[0] = *(const uint4*)w1;       b1.q[1] = *(const uint4*)(w1 + 8);

    c0 = wmma_bf16(a.v, b0.v, c0);
    c1 = wmma_bf16(a.v, b1.v, c1);
  }

  // C layout: vgpr i, lanes 0-15 -> row i, lanes 16-31 -> row i+8; col = lane&15
  float bb0 = bred[nl], bb1 = bred[16 + nl];
#pragma unroll
  for (int i = 0; i < 8; ++i) {
    int rg = m0 + wave * 16 + i + 8 * half;
    z[(size_t)rg * D_RED + nl]      = c0[i] + bb0;
    z[(size_t)rg * D_RED + nl + 16] = c1[i] + bb1;
  }
}

// ---------------------------------------------------------------------------
// Kernel 2: fused Plücker pair products + normalize + g = p_hat @ W_plu + b.
// Block = 256 threads (8 waves), 128-row M tile, N=1024 in 8 chunks of 128.
// LDS: p_hat 128x512 bf16 (128KB) + W_pluT chunk 128x512 bf16 (128KB) = 256KB.
// ---------------------------------------------------------------------------
__global__ __launch_bounds__(256) void k_plucker_gemm2(
    const float* __restrict__ z, const unsigned short* __restrict__ wpT,
    const float* __restrict__ bplu, const int* __restrict__ delta_p,
    float* __restrict__ out) {
  extern __shared__ unsigned char smem[];
  unsigned short* p_lds = (unsigned short*)smem;              // 131072 B
  unsigned short* b_lds = (unsigned short*)(smem + 131072);   // 131072 B
  float* zt_lds = (float*)(smem + 131072);                    // overlays b_lds
  float* zd_lds = (float*)(smem + 131072 + 16384);

  const int delta = *delta_p;
  const int tid  = threadIdx.x;
  const int wave = tid >> 5, lane = tid & 31;
  const int half = lane >> 4, nl = lane & 15;
  const int m0   = blockIdx.x * 128;

  // ---- Phase 1a: stage z rows (t and t-delta) into LDS
  for (int t = tid; t < 128 * 8; t += 256) {
    int r = t >> 3, cq = t & 7;
    int m = m0 + r;
    ((float4*)zt_lds)[t] = *(const float4*)&z[(size_t)m * D_RED + cq * 4];
    int md = m - delta; if (md < 0) md = 0;
    ((float4*)zd_lds)[t] = *(const float4*)&z[(size_t)md * D_RED + cq * 4];
  }
  __syncthreads();

  // ---- Phase 1b: pair products + norm; 2 threads per row, interleaved q
  {
    const int r = tid >> 1, c = tid & 1;
    const float* zt = &zt_lds[r * D_RED];
    const float* zd = &zd_lds[r * D_RED];
    float acc = 0.0f;
    int i = 0, j = 1 + c;                   // triu order: (0,1),(0,2),...
    for (int q = c; q < PAIRS; q += 2) {
      float p = zt[i] * zd[j] - zt[j] * zd[i];
      acc += p * p;
      p_lds[r * P_PAD + q] = f2bf(p);
      ++j; if (j == D_RED) { ++i; j = i + 1; }
      ++j; if (j == D_RED) { ++i; j = i + 1; }
    }
    acc += __shfl_xor(acc, 1, 32);          // pair lanes differ in bit 0
    float scale = 1.0f / fmaxf(sqrtf(acc), 1e-8f);
    for (int q = c; q < PAIRS; q += 2) {
      float pf = bf2f(p_lds[r * P_PAD + q]);
      p_lds[r * P_PAD + q] = f2bf(pf * scale);
    }
    for (int q = PAIRS + c; q < P_PAD; q += 2) p_lds[r * P_PAD + q] = 0;
  }

  // ---- Phase 2: GEMM over 8 N-chunks of 128 cols, K=512 from LDS
  for (int np = 0; np < 8; ++np) {
    __syncthreads();                        // prev chunk / phase 1 done
#if HAVE_TDM
    if (wave == 0) {
      // 128 rows of wpT (contiguous 128 KB): tile 512 x 128 of 2B elems
      tdm_load_2d(lds_offset(b_lds), wpT + (size_t)np * 128 * P_PAD,
                  /*2B*/1, /*tile*/P_PAD, 128, /*tensor*/P_PAD, D_MODEL,
                  /*stride0*/(unsigned long long)P_PAD);
      __builtin_amdgcn_s_wait_tensorcnt(0);
    }
#else
    for (int t = tid; t < 8192; t += 256)   // 8192 * 16B = 128 KB
      ((uint4*)b_lds)[t] = ((const uint4*)(wpT + (size_t)np * 128 * P_PAD))[t];
#endif
    __syncthreads();

    v8f acc[8] = {};
    for (int ks = 0; ks < 16; ++ks) {
      const int k0 = ks * 32;
      Frag a;
      const unsigned short* prow = &p_lds[(wave * 16 + nl) * P_PAD + k0];
      a.q[0] = *(const uint4*)(prow + half * 16);
      a.q[1] = *(const uint4*)(prow + half * 16 + 8);
#pragma unroll
      for (int nt = 0; nt < 8; ++nt) {
        Frag b;
        const unsigned short* brow = &b_lds[(nt * 16 + nl) * P_PAD + k0 + half * 16];
        b.q[0] = *(const uint4*)brow;
        b.q[1] = *(const uint4*)(brow + 8);
        acc[nt] = wmma_bf16(a.v, b.v, acc[nt]);
      }
    }

    // Epilogue: bias add; rows with (l < delta) are exactly zero
#pragma unroll
    for (int nt = 0; nt < 8; ++nt) {
      int col = np * 128 + nt * 16 + nl;
      float bias = bplu[col];
#pragma unroll
      for (int i = 0; i < 8; ++i) {
        int rg = m0 + wave * 16 + i + 8 * half;
        int l  = rg & (L_SEQ - 1);
        out[(size_t)rg * D_MODEL + col] = (l >= delta) ? (acc[nt][i] + bias) : 0.0f;
      }
    }
  }
}

// ---------------------------------------------------------------------------
extern "C" void kernel_launch(void* const* d_in, const int* in_sizes, int n_in,
                              void* d_out, int out_size, void* d_ws, size_t ws_size,
                              hipStream_t stream) {
  (void)in_sizes; (void)n_in; (void)out_size; (void)ws_size;
  const float* h       = (const float*)d_in[0];
  const int*   delta_p = (const int*)d_in[1];
  const float* wred_w  = (const float*)d_in[2];
  const float* wred_b  = (const float*)d_in[3];
  const float* wplu_w  = (const float*)d_in[4];
  const float* wplu_b  = (const float*)d_in[5];
  float* out = (float*)d_out;

  // workspace layout
  char* ws = (char*)d_ws;
  float*          zbuf  = (float*)ws;                                   // 4 MiB
  unsigned short* wredT = (unsigned short*)(ws + (size_t)4 * 1024 * 1024);       // 64 KiB
  unsigned short* wpT   = (unsigned short*)(ws + (size_t)4 * 1024 * 1024 + 65536); // 1 MiB

  k_prep<<<(D_MODEL * P_PAD + 255) / 256, 256, 0, stream>>>(wred_w, wplu_w, wredT, wpT);
  k_gemm1<<<M_TOT / 128, 256, 0, stream>>>(h, wredT, wred_b, zbuf);
  k_plucker_gemm2<<<M_TOT / 128, 256, 262144, stream>>>(zbuf, wpT, wplu_b, delta_p, out);
}
